// KPCN_v1_8100308320532
// MI455X (gfx1250) — matrix operand order repaired and checked by reading, not compiled, not run on hardware
//
#include <hip/hip_runtime.h>
#include <hip/hip_bf16.h>

typedef __attribute__((ext_vector_type(16))) _Float16 v16h;
typedef __attribute__((ext_vector_type(8)))  _Float16 v8h;
typedef __attribute__((ext_vector_type(8)))  float    v8f;
typedef __attribute__((ext_vector_type(4)))  unsigned int u32x4;
typedef __attribute__((ext_vector_type(8)))  int     i32x8;
typedef __attribute__((ext_vector_type(4)))  int     i32x4;

// ---------------- geometry ----------------
#define HH 512
#define WW 1024
#define WP 1026                 // padded width  (H pad -> 514 rows)
#define CPAD 128                // padded channel count (f16)
#define NT7 7                   // 112 output channels for feature layers
#define NT28 28                 // 448 output channels for last layer
#define KREG 21
#define PADR 10
#define HR (HH + 2*PADR)        // 532
#define WR (WW + 2*PADR)        // 1044

static constexpr size_t ACT_BYTES  = (size_t)514 * WP * CPAD * 2;      // 135,005,184
static constexpr size_t RGBP_BYTES = (size_t)3 * HR * WR * 4;          // 6,664,896
static constexpr long   S7H  = 9L * 4 * NT7  * 32 * 16;                // 129,024 halfs / layer
static constexpr long   S28H = 9L * 4 * NT28 * 32 * 16;                // 516,096 halfs
static constexpr long   WPACK_TOTAL = 24L * S7H + S28H;                // 3,612,672 halfs
static constexpr int    NBIAS = 24 * 112 + 448;                        // 3136

static constexpr size_t OFF_ACTA  = 0;
static constexpr size_t OFF_ACTB  = ACT_BYTES;
static constexpr size_t OFF_RGBP  = 2 * ACT_BYTES;                     // 270,010,368
static constexpr size_t OFF_WPACK = OFF_RGBP + RGBP_BYTES;             // 276,675,264 (16B aligned)
static constexpr size_t OFF_BIAS  = OFF_WPACK + (size_t)WPACK_TOTAL*2; // 283,900,608

// ---------------- TDM: async tensor load (global -> LDS) ----------------
// Builds a D# per CDNA5 ISA ch.8: group0 {count=1 | lds_addr | global_addr | type=2},
// group1 {data_size=8B, tensor_dim0/1, tile_dim0/1, dim0_stride}, groups 2/3
// degenerate (tile_dim2/3/4 = 0 -> unused). Units of 8 bytes (data_size=3).
// Tracked by TENSORcnt; waited with s_wait_tensorcnt.
__device__ __forceinline__ void tdm_load_to_lds(const void* gsrc, void* ldst,
                                                unsigned tile_d0_u8,   // row length, 8B units (<65536)
                                                unsigned tile_d1,      // number of rows (<65536)
                                                unsigned stride_u8) {  // row stride, 8B units
  unsigned long long ga = (unsigned long long)(uintptr_t)gsrc;
  unsigned la = (unsigned)(uintptr_t)ldst;   // flat LDS ptr: low 32 bits = LDS byte offset
  u32x4 g0;
  g0[0] = 1u;                                            // count=1, user descriptor
  g0[1] = la;                                            // lds_addr
  g0[2] = (unsigned)ga;                                  // global_addr[31:0]
  g0[3] = (unsigned)((ga >> 32) & 0x01FFFFFFu) | 0x80000000u; // global_addr[56:32] | type=2
  i32x8 g1;
  g1[0] = (int)(3u << 16);                               // wg_mask=0, data_size=3 (8B)
  g1[1] = (int)((tile_d0_u8 & 0xFFFFu) << 16);           // tensor_dim0[15:0]
  g1[2] = (int)((tile_d1 & 0xFFFFu) << 16);              // tensor_dim0[31:16]=0, tensor_dim1[15:0]
  g1[3] = (int)((tile_d0_u8 & 0xFFFFu) << 16);           // tensor_dim1[31:16]=0, tile_dim0
  g1[4] = (int)(tile_d1 & 0xFFFFu);                      // tile_dim1, tile_dim2=0 (unused)
  g1[5] = (int)stride_u8;                                // tensor_dim0_stride[31:0]
  g1[6] = 0;                                             // stride hi, dim1_stride lo
  g1[7] = 0;
  i32x4 g2; g2[0] = 1; g2[1] = 1; g2[2] = 0; g2[3] = 0;  // tensor_dim2/3=1, tile_dim3=0
  i32x4 g3; g3[0] = 0; g3[1] = (int)(1u << 16); g3[2] = 0; g3[3] = 0; // tensor_dim4=1, tile_dim4=0
#if __clang_major__ >= 23
  i32x8 z8 = {};
  __builtin_amdgcn_tensor_load_to_lds(g0, g1, g2, g3, z8, 0);
#else
  __builtin_amdgcn_tensor_load_to_lds(g0, g1, g2, g3, 0);
#endif
}

// ---------------- init / packing ----------------
__global__ void zero_kernel(uint4* p, long n16) {
  uint4 z; z.x = z.y = z.z = z.w = 0u;
  for (long i = blockIdx.x * (long)blockDim.x + threadIdx.x; i < n16;
       i += (long)gridDim.x * blockDim.x)
    p[i] = z;
}

__global__ void pack_in_kernel(const float* __restrict__ in,
                               _Float16* __restrict__ act,
                               float* __restrict__ rgbp) {
  int i = blockIdx.x * blockDim.x + threadIdx.x;
  if (i >= HH * WW) return;
  int y = i >> 10;
  int x = i & 1023;
  _Float16* a = act + ((size_t)(y + 1) * WP + (x + 1)) * CPAD;
#pragma unroll
  for (int c = 0; c < 10; ++c)
    a[c] = (_Float16)in[((size_t)c * HH + y) * WW + x];
#pragma unroll
  for (int c = 0; c < 3; ++c)
    rgbp[((size_t)c * HR + (y + PADR)) * WR + (x + PADR)] =
        in[((size_t)c * HH + y) * WW + x];
}

// dest layout per layer: idx = ((((s*4+kt)*NT + nt)*32 + lane)*16 + e)
// B-fragment lane mapping: cin = kt*32 + (lane>=16?8:0) + (e<8 ? e : e+8) ; cout = nt*16 + (lane&15)
__global__ void pack_w_kernel(const float* __restrict__ w0,
                              const float* __restrict__ wmid,
                              const float* __restrict__ wlast,
                              _Float16* __restrict__ wp) {
  for (long idx = blockIdx.x * (long)blockDim.x + threadIdx.x; idx < WPACK_TOTAL;
       idx += (long)gridDim.x * blockDim.x) {
    long rem; int li, NT;
    if (idx < 24L * S7H) { li = (int)(idx / S7H); rem = idx % S7H; NT = NT7; }
    else                 { li = 24; rem = idx - 24L * S7H; NT = NT28; }
    int e    = (int)(rem & 15);  long t = rem >> 4;
    int lane = (int)(t & 31);    t >>= 5;
    int nt   = (int)(t % NT);    t /= NT;
    int kt   = (int)(t & 3);     t >>= 2;
    int s    = (int)t;
    int cin  = kt * 32 + ((lane >= 16) ? 8 : 0) + ((e < 8) ? e : e + 8);
    int cout = nt * 16 + (lane & 15);
    int dy = s / 3, dx = s % 3;
    float val = 0.f;
    if (li == 0) {
      if (cin < 10 && cout < 100)
        val = w0[((cout * 10 + cin) * 3 + dy) * 3 + dx];
    } else if (li <= 23) {
      int m = li - 1;
      if (cin < 100 && cout < 100)
        val = wmid[((((long)m * 100 + cout) * 100 + cin) * 3 + dy) * 3 + dx];
    } else {
      if (cin < 100 && cout < 441)
        val = wlast[(((long)cout * 100 + cin) * 3 + dy) * 3 + dx];
    }
    wp[idx] = (_Float16)val;
  }
}

__global__ void pack_b_kernel(const float* __restrict__ b0,
                              const float* __restrict__ bmid,
                              const float* __restrict__ blast,
                              float* __restrict__ bp) {
  int i = blockIdx.x * blockDim.x + threadIdx.x;
  if (i >= NBIAS) return;
  float v = 0.f;
  if (i < 112)            { if (i < 100) v = b0[i]; }
  else if (i < 24 * 112)  { int j = i - 112; int m = j / 112, c = j % 112;
                            if (c < 100) v = bmid[m * 100 + c]; }
  else                    { int c = i - 24 * 112; if (c < 441) v = blast[c]; }
  bp[i] = v;
}

// ---------------- WMMA conv layer: implicit GEMM, 9 shifts x (ktiles) k-steps ----------------
#define SW_HALFS (4 * NT7 * 32 * 16)   // 14336 halfs = 28672 B per shift

__global__ __launch_bounds__(256) void conv_gemm(
    const _Float16* __restrict__ in, _Float16* __restrict__ out,
    const _Float16* __restrict__ w, const float* __restrict__ bias,
    int ktiles, int leaky) {
  __shared__ _Float16 sw[2][SW_HALFS];       // double-buffered weight panels (2 x 28 KB)
  const int tid  = threadIdx.x;
  const int lane = tid & 31;
  const int wave = tid >> 5;
  const int tile = blockIdx.x * 8 + wave;     // 32768 tiles total
  const int y  = tile >> 6;                   // row
  const int x0 = (tile & 63) << 4;            // 16-pixel tile start
  const int lh = lane >> 4;
  const int ln = lane & 15;

  v8f acc[NT7];
#pragma unroll
  for (int nt = 0; nt < NT7; ++nt) {
    float b = bias[nt * 16 + ln];             // D: n = lane&15
#pragma unroll
    for (int r = 0; r < 8; ++r) acc[nt][r] = b;
  }

  // wave 0 drives one TDM per weight panel for the whole workgroup
  if (wave == 0)
    tdm_load_to_lds(w, &sw[0][0], (SW_HALFS * 2) / 8, 1, (SW_HALFS * 2) / 8);

  for (int s = 0; s < 9; ++s) {
    const int buf = s & 1;
    if (wave == 0) __builtin_amdgcn_s_wait_tensorcnt(0);  // panel s landed in LDS
    __syncthreads();                                      // all waves see it; prev panel free
    if (wave == 0 && s < 8)                               // overlap next DMA with compute
      tdm_load_to_lds(w + (size_t)(s + 1) * SW_HALFS, &sw[buf ^ 1][0],
                      (SW_HALFS * 2) / 8, 1, (SW_HALFS * 2) / 8);

    const int dy = s / 3, dx = s % 3;
    // A row m = lane&15 -> padded pixel (y+dy, x0+m+dx)
    const _Float16* arow = in + ((size_t)(y + dy) * WP + (x0 + ln + dx)) * CPAD;
    for (int kt = 0; kt < ktiles; ++kt) {
      const int ch0 = kt * 32 + lh * 8;       // A chunks: K = ch0..ch0+7, ch0+16..ch0+23
      v8h a0 = *(const v8h*)(arow + ch0);
      v8h a1 = *(const v8h*)(arow + ch0 + 16);
      v16h a;
#pragma unroll
      for (int i = 0; i < 8; ++i) { a[i] = a0[i]; a[i + 8] = a1[i]; }
#pragma unroll
      for (int nt = 0; nt < NT7; ++nt) {
        v16h b = *(const v16h*)(&sw[buf][0] + ((size_t)(kt * NT7 + nt) * 32 + lane) * 16);
        acc[nt] = __builtin_amdgcn_wmma_f32_16x16x32_f16(
            false, a, false, b, (short)0, acc[nt], false, false);
      }
    }
  }

  // epilogue: D element (lane, r) -> pixel m = r+8*lh, channel n = nt*16 + ln
  _Float16* orow = out + ((size_t)(y + 1) * WP + (x0 + 1)) * CPAD;
#pragma unroll
  for (int nt = 0; nt < NT7; ++nt) {
#pragma unroll
    for (int r = 0; r < 8; ++r) {
      float v = acc[nt][r];
      if (leaky) v = v > 0.f ? v : 0.01f * v;
      int m = r + 8 * lh;
      orow[(size_t)m * CPAD + nt * 16 + ln] = (_Float16)v;
    }
  }
}

// ---------------- fused last layer: 441 kernel weights -> normalize -> regression ----------------
__global__ __launch_bounds__(64) void last_fused(
    const _Float16* __restrict__ in, const _Float16* __restrict__ w,
    const float* __restrict__ bias, const float* __restrict__ rgbp,
    float* __restrict__ outp) {
  __shared__ _Float16 sact[2][3 * 18 * CPAD];   // 2 x 13824 B input patch
  __shared__ _Float16 swt[2][16][448];          // 2 x 14336 B predicted weights
  __shared__ float    ssum[2][16];

  const int tid  = threadIdx.x;
  const int lane = tid & 31;
  const int wave = tid >> 5;
  const int tile = blockIdx.x * 2 + wave;
  const int y  = tile >> 6;
  const int x0 = (tile & 63) << 4;
  const int lh = lane >> 4;
  const int ln = lane & 15;

  // TDM 2D tile: 3 rows x 4608 B (18 px x 128 ch f16), row stride WP*256 B
  tdm_load_to_lds(in + ((size_t)y * WP + x0) * CPAD, &sact[wave][0],
                  (18 * CPAD * 2) / 8, 3, (WP * CPAD * 2) / 8);
  __builtin_amdgcn_s_wait_tensorcnt(0);
  __syncthreads();

  float psum[8];
#pragma unroll
  for (int r = 0; r < 8; ++r) psum[r] = 0.f;

  for (int nt = 0; nt < NT28; ++nt) {
    v8f acc;
    float bb = bias[nt * 16 + ln];
#pragma unroll
    for (int r = 0; r < 8; ++r) acc[r] = bb;
    for (int s = 0; s < 9; ++s) {
      const int dy = s / 3, dx = s % 3;
      const _Float16* aw = sact[wave] + ((size_t)(dy * 18) + (ln + dx)) * CPAD;
#pragma unroll
      for (int kt = 0; kt < 4; ++kt) {
        const int ch0 = kt * 32 + lh * 8;
        v8h a0 = *(const v8h*)(aw + ch0);
        v8h a1 = *(const v8h*)(aw + ch0 + 16);
        v16h a;
#pragma unroll
        for (int i = 0; i < 8; ++i) { a[i] = a0[i]; a[i + 8] = a1[i]; }
        v16h b = *(const v16h*)(w + ((((size_t)s * 4 + kt) * NT28 + nt) * 32 + lane) * 16);
        acc = __builtin_amdgcn_wmma_f32_16x16x32_f16(
            false, a, false, b, (short)0, acc, false, false);
      }
    }
    // per-pixel sum (reduce over N within each 16-lane half) + stash weights
#pragma unroll
    for (int r = 0; r < 8; ++r) {
      float v = acc[r];
      float t = v;
      t += __shfl_xor(t, 1);
      t += __shfl_xor(t, 2);
      t += __shfl_xor(t, 4);
      t += __shfl_xor(t, 8);
      psum[r] += t;                             // pixel m = r + 8*lh
      swt[wave][r + 8 * lh][nt * 16 + ln] = (_Float16)v;
    }
  }
  if (ln == 0) {
#pragma unroll
    for (int r = 0; r < 8; ++r) ssum[wave][r + 8 * lh] = psum[r];
  }
  __syncthreads();

  // normalization + 441-tap kernel regression (half the taps per lane-half)
  const int p = ln;
  const int x = x0 + p;
  const float inv = 1.0f / (ssum[wave][p] + 1e-6f);
  float r0 = 0.f, r1 = 0.f, r2 = 0.f;
  const int kbeg = lh * 224;
  const int kend = kbeg + 224;
  for (int k = kbeg; k < kend; ++k) {
    if (k >= KREG * KREG) break;                // padded entries are zero anyway
    float wn = (float)swt[wave][p][k] * inv;
    int dy = k / KREG, dx = k % KREG;
    size_t base = (size_t)(y + dy) * WR + (x + dx);
    r0 += wn * rgbp[base];
    r1 += wn * rgbp[(size_t)HR * WR + base];
    r2 += wn * rgbp[2 * (size_t)HR * WR + base];
  }
  r0 += __shfl_xor(r0, 16);
  r1 += __shfl_xor(r1, 16);
  r2 += __shfl_xor(r2, 16);
  if (lh == 0) {
    size_t o = (size_t)y * WW + x;
    outp[o] = r0;
    outp[(size_t)HH * WW + o] = r1;
    outp[2 * (size_t)HH * WW + o] = r2;
  }
}

// ---------------- host ----------------
extern "C" void kernel_launch(void* const* d_in, const int* in_sizes, int n_in,
                              void* d_out, int out_size, void* d_ws, size_t ws_size,
                              hipStream_t stream) {
  const float* input = (const float*)d_in[0];
  // d_in[1] = ref (unused by reference output)
  const float* w0    = (const float*)d_in[2];
  const float* b0    = (const float*)d_in[3];
  const float* wmid  = (const float*)d_in[4];
  const float* bmid  = (const float*)d_in[5];
  const float* wlast = (const float*)d_in[6];
  const float* blast = (const float*)d_in[7];
  float* out = (float*)d_out;

  char* ws = (char*)d_ws;
  _Float16* actA = (_Float16*)(ws + OFF_ACTA);
  _Float16* actB = (_Float16*)(ws + OFF_ACTB);
  float*    rgbp = (float*)(ws + OFF_RGBP);
  _Float16* wp   = (_Float16*)(ws + OFF_WPACK);
  float*    bp   = (float*)(ws + OFF_BIAS);

  // zero activation ping-pong buffers + rgb pad plane (borders / pad channels)
  zero_kernel<<<4096, 256, 0, stream>>>((uint4*)ws, (long)(OFF_WPACK / 16));
  pack_in_kernel<<<(HH * WW) / 256, 256, 0, stream>>>(input, actA, rgbp);
  pack_w_kernel<<<4096, 256, 0, stream>>>(w0, wmid, wlast, wp);
  pack_b_kernel<<<(NBIAS + 255) / 256, 256, 0, stream>>>(b0, bmid, blast, bp);

  // conv0: 10 -> 100, single k-step (Cin padded to 32), no activation
  conv_gemm<<<4096, 256, 0, stream>>>(actA, actB, wp, bp, 1, 0);
  _Float16* cur = actB;
  _Float16* nxt = actA;
  for (int m = 0; m < 23; ++m) {
    conv_gemm<<<4096, 256, 0, stream>>>(cur, nxt, wp + (size_t)(1 + m) * S7H,
                                        bp + (1 + m) * 112, 4, 1);
    _Float16* t = cur; cur = nxt; nxt = t;
  }
  // last conv (100 -> 441) fused with normalization + kernel regression
  last_fused<<<32768 / 2, 64, 0, stream>>>(cur, wp + 24L * S7H, bp + 24 * 112,
                                           rgbp, out);
}
